// Attention_50998441672798
// MI455X (gfx1250) — matrix operand order repaired and checked by reading, not compiled, not run on hardware
//
#include <hip/hip_runtime.h>
#include <hip/hip_bf16.h>

typedef __bf16 bf16;
typedef __attribute__((ext_vector_type(16))) __bf16 v16bf;
typedef __attribute__((ext_vector_type(8)))  __bf16 v8bf;
typedef __attribute__((ext_vector_type(4)))  __bf16 v4bf;
typedef __attribute__((ext_vector_type(8)))  float  v8f;
typedef __attribute__((ext_vector_type(4)))  int    v4i;

#define GLOBAL_AS __attribute__((address_space(1)))
#define LDS_AS    __attribute__((address_space(3)))

#if __has_builtin(__builtin_amdgcn_global_load_async_to_lds_b128) && \
    __has_builtin(__builtin_amdgcn_s_wait_asynccnt)
#define HAVE_ASYNC_LDS 1
#else
#define HAVE_ASYNC_LDS 0
#endif

__device__ __forceinline__ v8f wmma_bf16(v16bf a, v16bf b, v8f c) {
  return __builtin_amdgcn_wmma_f32_16x16x32_bf16(
      false, a, false, b, (short)0, c, false, false);
}

// A fragment (16x32, MxK) from row-major [16][stride] bf16 tile.
__device__ __forceinline__ v16bf load_frag_A(const bf16* tile, int stride, int lane) {
  int kg = lane >> 4;
  const bf16* p = tile + (lane & 15) * stride;
  v16bf a;
#pragma unroll
  for (int j = 0; j < 16; ++j) {
    int k = (j < 8) ? (kg * 8 + j) : (16 + kg * 8 + (j - 8));
    a[j] = p[k];
  }
  return a;
}

// B fragment (32x16, KxN) from N-major/K-contiguous tile Bt[n][k]; table k order (matches A).
__device__ __forceinline__ v16bf load_frag_B(const bf16* tileT, int n, int stride,
                                             int kbase, int lane) {
  const bf16* p = tileT + n * stride + kbase;
  int kg = lane >> 4;
  v16bf b;
#pragma unroll
  for (int j = 0; j < 16; ++j) b[j] = p[(j < 8) ? (kg * 8 + j) : (16 + kg * 8 + (j - 8))];
  return b;
}

__device__ __forceinline__ void cvt_store4(bf16* dst, float4 v) {
  v4bf o;
  o[0] = (bf16)v.x; o[1] = (bf16)v.y; o[2] = (bf16)v.z; o[3] = (bf16)v.w;
  *(v4bf*)dst = o;
}

// ---------------- Kernel 1: QKV projection + RoPE ----------------
__global__ __launch_bounds__(256) void qkv_rope_kernel(
    const float* __restrict__ x, const float* __restrict__ rot,
    const float* __restrict__ w_qkv,
    bf16* __restrict__ qbuf, bf16* __restrict__ kbuf, bf16* __restrict__ vbuf) {
  __shared__ bf16 ldsA[2][128 * 32];
  __shared__ bf16 ldsB[2][64 * 32];
  const int t = threadIdx.x, lane = t & 31, wave = t >> 5;
  const int rowBase = blockIdx.x * 128;
  const int cb = blockIdx.y;            // type*16 + head
  const int colBase = cb * 64;
  v8f c[4] = {};
  float4 ra[4], rb[2];

  auto load_tiles = [&](int k0) {
#pragma unroll
    for (int i = 0; i < 4; ++i) {       // A: 128x32 f32 = 1024 float4
      int e = t + i * 256;
      ra[i] = *(const float4*)&x[(size_t)(rowBase + (e >> 3)) * 1024 + k0 + (e & 7) * 4];
    }
#pragma unroll
    for (int i = 0; i < 2; ++i) {       // B: 64x32 f32 = 512 float4
      int e = t + i * 256;
      rb[i] = *(const float4*)&w_qkv[(size_t)(colBase + (e >> 3)) * 1024 + k0 + (e & 7) * 4];
    }
  };
  auto store_tiles = [&](int buf) {
#pragma unroll
    for (int i = 0; i < 4; ++i) {
      int e = t + i * 256;
      cvt_store4(&ldsA[buf][(e >> 3) * 32 + (e & 7) * 4], ra[i]);
    }
#pragma unroll
    for (int i = 0; i < 2; ++i) {
      int e = t + i * 256;
      cvt_store4(&ldsB[buf][(e >> 3) * 32 + (e & 7) * 4], rb[i]);
    }
  };

  load_tiles(0);
  store_tiles(0);
  for (int kk = 0; kk < 32; ++kk) {
    const int cur = kk & 1;
    if (kk + 1 < 32) load_tiles((kk + 1) * 32);   // overlap with barrier+compute
    __syncthreads();
    v16bf a = load_frag_A(&ldsA[cur][wave * 16 * 32], 32, lane);
#pragma unroll
    for (int f = 0; f < 4; ++f) {
      v16bf b = load_frag_B(&ldsB[cur][0], 16 * f + (lane & 15), 32, 0, lane);
      c[f] = wmma_bf16(a, b, c[f]);
    }
    if (kk + 1 < 32) store_tiles(cur ^ 1);
  }

  // RoPE epilogue (partner d+-32 == fragment f+-2, same lane/reg)
  const int type = cb >> 4, head = cb & 15;
  bf16* dst = (type == 0) ? qbuf : (type == 1) ? kbuf : vbuf;
  const int kg = lane >> 4;
#pragma unroll
  for (int r = 0; r < 8; ++r) {
    int gm = rowBase + wave * 16 + kg * 8 + r;
    int bidx = gm >> 11;
    int n = gm & 2047;
#pragma unroll
    for (int f = 0; f < 4; ++f) {
      int d = 16 * f + (lane & 15);
      float val = c[f][r];
      if (type < 2) {
        float p = rot[n * 64 + d];
        float other = (f < 2) ? -c[f + 2][r] : c[f - 2][r];
        val = val * __cosf(p) + other * __sinf(p);
      }
      dst[(((size_t)(bidx * 16 + head) * 2048 + n) << 6) + d] = (bf16)val;
    }
  }
}

// ---------------- Kernel 2: flash attention ----------------
__global__ __launch_bounds__(256) void flash_attn_kernel(
    const bf16* __restrict__ qbuf, const bf16* __restrict__ kbuf,
    const bf16* __restrict__ vbuf, bf16* __restrict__ obuf) {
  __shared__ bf16 ldsK[2][64 * 64];    // [key][d]
  __shared__ bf16 ldsVT[2][64 * 64];   // [d][key]
  __shared__ bf16 ldsP[8][16 * 64];
  const int t = threadIdx.x, lane = t & 31, wave = t >> 5;
  const int head = blockIdx.y, batch = blockIdx.z;
  const size_t hb = (size_t)(batch * 16 + head) * 2048 * 64;
  const bf16* Q = qbuf + hb;
  const bf16* K = kbuf + hb;
  const bf16* V = vbuf + hb;
  const int q0 = blockIdx.x * 128 + wave * 16;
  const int kg = lane >> 4;

  v8bf rv[2];
#if !HAVE_ASYNC_LDS
  v8bf rk[2];
#endif
  // 64x64 bf16 tile = 512 chunks of 8 bf16; 256 threads -> 2 each.
  auto load_v_regs = [&](int kt) {
#pragma unroll
    for (int i = 0; i < 2; ++i) {
      int e = t + i * 256;
      size_t g = (size_t)(kt * 64 + (e >> 3)) * 64 + (e & 7) * 8;
      rv[i] = *(const v8bf*)&V[g];
#if !HAVE_ASYNC_LDS
      rk[i] = *(const v8bf*)&K[g];
#endif
    }
  };
  auto issue_k_async = [&](int kt, int buf) {
#if HAVE_ASYNC_LDS
#pragma unroll
    for (int i = 0; i < 2; ++i) {
      int e = t + i * 256;
      int key = e >> 3, c8 = (e & 7) * 8;
      __builtin_amdgcn_global_load_async_to_lds_b128(
          (GLOBAL_AS v4i*)&K[(size_t)(kt * 64 + key) * 64 + c8],
          (LDS_AS v4i*)&ldsK[buf][key * 64 + c8], 0, 0);
    }
#else
    (void)kt; (void)buf;
#endif
  };
  auto store_regs = [&](int buf) {
#pragma unroll
    for (int i = 0; i < 2; ++i) {
      int e = t + i * 256;
      int key = e >> 3, c8 = (e & 7) * 8;
#if !HAVE_ASYNC_LDS
      *(v8bf*)&ldsK[buf][key * 64 + c8] = rk[i];
#endif
#pragma unroll
      for (int j = 0; j < 8; ++j)
        ldsVT[buf][(c8 + j) * 64 + key] = rv[i][j];   // transpose V
    }
  };

  // Q fragments held in registers for the whole kernel
  v16bf qa[2];
  {
    const int qrow = q0 + (lane & 15);
#pragma unroll
    for (int ks = 0; ks < 2; ++ks)
#pragma unroll
      for (int j = 0; j < 16; ++j) {
        int k = ks * 32 + ((j < 8) ? (kg * 8 + j) : (16 + kg * 8 + (j - 8)));
        qa[ks][j] = Q[(size_t)qrow * 64 + k];
      }
  }

  v8f o[4] = {};
  float mrow[8], lrow[8];
#pragma unroll
  for (int r = 0; r < 8; ++r) { mrow[r] = -1e30f; lrow[r] = 0.f; }

  // prologue: tile 0 -> buffer 0
  load_v_regs(0);
  issue_k_async(0, 0);
  store_regs(0);

  for (int kt = 0; kt < 32; ++kt) {
    const int cur = kt & 1;
    const bool more = (kt + 1 < 32);
    if (more) load_v_regs(kt + 1);
#if HAVE_ASYNC_LDS
    __builtin_amdgcn_s_wait_asynccnt(0);   // this thread's K-tile DMA done
#endif
    __syncthreads();

    // S = Q K^T
    v8f s[4] = {};
#pragma unroll
    for (int ks = 0; ks < 2; ++ks)
#pragma unroll
      for (int f = 0; f < 4; ++f) {
        v16bf bk = load_frag_B(&ldsK[cur][0], 16 * f + (lane & 15), 64, ks * 32, lane);
        s[f] = wmma_bf16(qa[ks], bk, s[f]);
      }

    // online softmax (rows split across 16-lane halves)
    float p[4][8];
#pragma unroll
    for (int r = 0; r < 8; ++r) {
      float mx = -1e30f;
#pragma unroll
      for (int f = 0; f < 4; ++f) mx = fmaxf(mx, s[f][r] * 0.125f);
#pragma unroll
      for (int msk = 1; msk < 16; msk <<= 1) mx = fmaxf(mx, __shfl_xor(mx, msk));
      float mnew = fmaxf(mrow[r], mx);
      float alpha = __expf(mrow[r] - mnew);
      float rs = 0.f;
#pragma unroll
      for (int f = 0; f < 4; ++f) {
        float e_ = __expf(s[f][r] * 0.125f - mnew);
        p[f][r] = e_;
        rs += e_;
      }
#pragma unroll
      for (int msk = 1; msk < 16; msk <<= 1) rs += __shfl_xor(rs, msk);
      lrow[r] = lrow[r] * alpha + rs;
      mrow[r] = mnew;
#pragma unroll
      for (int f = 0; f < 4; ++f) o[f][r] *= alpha;
    }

    // C-layout -> A-layout transpose of P through per-wave LDS
    bf16* myP = &ldsP[wave][0];
#pragma unroll
    for (int r = 0; r < 8; ++r) {
      int row = kg * 8 + r;
#pragma unroll
      for (int f = 0; f < 4; ++f)
        myP[row * 64 + 16 * f + (lane & 15)] = (bf16)p[f][r];
    }
    asm volatile("s_wait_dscnt 0" ::: "memory");

    // O += P V
#pragma unroll
    for (int ks = 0; ks < 2; ++ks) {
      v16bf pa;
#pragma unroll
      for (int j = 0; j < 16; ++j) {
        int k = ks * 32 + ((j < 8) ? (kg * 8 + j) : (16 + kg * 8 + (j - 8)));
        pa[j] = myP[(lane & 15) * 64 + k];
      }
#pragma unroll
      for (int f = 0; f < 4; ++f) {
        v16bf bv = load_frag_B(&ldsVT[cur][0], 16 * f + (lane & 15), 64, ks * 32, lane);
        o[f] = wmma_bf16(pa, bv, o[f]);
      }
    }

    if (more) { issue_k_async(kt + 1, cur ^ 1); store_regs(cur ^ 1); }
  }

#pragma unroll
  for (int r = 0; r < 8; ++r) {
    float inv = 1.0f / lrow[r];
    int n = q0 + kg * 8 + r;
#pragma unroll
    for (int f = 0; f < 4; ++f) {
      int col = head * 64 + 16 * f + (lane & 15);
      obuf[(size_t)(batch * 2048 + n) * 1024 + col] = (bf16)(o[f][r] * inv);
    }
  }
}

// ---------------- Kernel 3: output projection ----------------
__global__ __launch_bounds__(256) void out_proj_kernel(
    const bf16* __restrict__ attn, const float* __restrict__ w_out,
    float* __restrict__ out) {
  __shared__ bf16 ldsA[2][128 * 32];
  __shared__ bf16 ldsB[2][64 * 32];
  const int t = threadIdx.x, lane = t & 31, wave = t >> 5;
  const int rowBase = blockIdx.x * 128;
  const int colBase = blockIdx.y * 64;
  v8f c[4] = {};
  v8bf raA[2];
  float4 rbB[2];

  auto load_tiles = [&](int k0) {
#pragma unroll
    for (int i = 0; i < 2; ++i) {       // A: 128x32 bf16 = 512 chunks of 8
      int e = t + i * 256;
      raA[i] = *(const v8bf*)&attn[(size_t)(rowBase + (e >> 2)) * 1024 + k0 + (e & 3) * 8];
    }
#pragma unroll
    for (int i = 0; i < 2; ++i) {       // B: 64x32 f32 = 512 float4
      int e = t + i * 256;
      rbB[i] = *(const float4*)&w_out[(size_t)(colBase + (e >> 3)) * 1024 + k0 + (e & 7) * 4];
    }
  };
  auto store_tiles = [&](int buf) {
#pragma unroll
    for (int i = 0; i < 2; ++i) {
      int e = t + i * 256;
      *(v8bf*)&ldsA[buf][(e >> 2) * 32 + (e & 3) * 8] = raA[i];
    }
#pragma unroll
    for (int i = 0; i < 2; ++i) {
      int e = t + i * 256;
      cvt_store4(&ldsB[buf][(e >> 3) * 32 + (e & 7) * 4], rbB[i]);
    }
  };

  load_tiles(0);
  store_tiles(0);
  for (int kk = 0; kk < 32; ++kk) {
    const int cur = kk & 1;
    if (kk + 1 < 32) load_tiles((kk + 1) * 32);
    __syncthreads();
    v16bf a = load_frag_A(&ldsA[cur][wave * 16 * 32], 32, lane);
#pragma unroll
    for (int f = 0; f < 4; ++f) {
      v16bf b = load_frag_B(&ldsB[cur][0], 16 * f + (lane & 15), 32, 0, lane);
      c[f] = wmma_bf16(a, b, c[f]);
    }
    if (kk + 1 < 32) store_tiles(cur ^ 1);
  }
  const int kg = lane >> 4;
#pragma unroll
  for (int r = 0; r < 8; ++r) {
    size_t gm = rowBase + wave * 16 + kg * 8 + r;
#pragma unroll
    for (int f = 0; f < 4; ++f)
      out[gm * 1024 + colBase + 16 * f + (lane & 15)] = c[f][r];
  }
}

extern "C" void kernel_launch(void* const* d_in, const int* in_sizes, int n_in,
                              void* d_out, int out_size, void* d_ws, size_t ws_size,
                              hipStream_t stream) {
  (void)in_sizes; (void)n_in; (void)out_size; (void)ws_size;
  const float* x     = (const float*)d_in[0];
  const float* rot   = (const float*)d_in[1];
  const float* w_qkv = (const float*)d_in[2];
  const float* w_out = (const float*)d_in[3];
  float* out = (float*)d_out;

  const size_t QELEMS = (size_t)4 * 16 * 2048 * 64;
  char* ws = (char*)d_ws;
  bf16* qb = (bf16*)ws;
  bf16* kb = (bf16*)(ws + QELEMS * 2);
  bf16* vb = (bf16*)(ws + QELEMS * 4);
  bf16* ob = (bf16*)(ws + QELEMS * 6);

  qkv_rope_kernel<<<dim3(64, 48), 256, 0, stream>>>(x, rot, w_qkv, qb, kb, vb);
  flash_attn_kernel<<<dim3(16, 16, 4), 256, 0, stream>>>(qb, kb, vb, ob);
  out_proj_kernel<<<dim3(64, 16), 256, 0, stream>>>(ob, w_out, out);
}
